// EmbedGCN_84327387889737
// MI455X (gfx1250) — matrix operand order repaired and checked by reading, not compiled, not run on hardware
//
#include <hip/hip_runtime.h>
#include <hip/hip_bf16.h>

#define N_NODES 4096
#define EDIM    64
#define DIN     256
#define DOUT    256
#define BATCH   32

typedef __attribute__((ext_vector_type(2)))  float   v2f;
typedef __attribute__((ext_vector_type(8)))  float   v8f;
typedef unsigned short u16;
typedef __attribute__((ext_vector_type(8)))  u16     v8u;
typedef __attribute__((ext_vector_type(16))) __bf16  v16bf;

// ---------- helpers: fp32 <-> bf16 (round-to-nearest-even), split hi/lo ----------
__device__ __forceinline__ u16 f2bf(float f) {
    unsigned int u = __float_as_uint(f);
    u = u + 0x7FFFu + ((u >> 16) & 1u);   // RNE
    return (u16)(u >> 16);
}
__device__ __forceinline__ float bf2f(u16 h) {
    return __uint_as_float(((unsigned int)h) << 16);
}
// Load a 16-half fragment as two b128 chunks (chunk1 at +second_off halves)
__device__ __forceinline__ v16bf ldfrag(const u16* __restrict__ p, int second_off) {
    v8u a0 = *(const v8u*)p;
    v8u a1 = *(const v8u*)(p + second_off);
    return __builtin_bit_cast(v16bf, __builtin_shufflevector(
        a0, a1, 0,1,2,3,4,5,6,7,8,9,10,11,12,13,14,15));
}

// ---------- 16x16 logits tile via exact f32 WMMA (K=64 in steps of 4) ----------
// f32 A 16x4 layout: lane M = lane%16, element j -> K = 2*(lane/16)+j  (mirrored for B)
__device__ __forceinline__ v8f logits_tile(const float* __restrict__ emb,
                                           int n0, int m0, int l16, int hl) {
    v8f c = {};
    const float* arow = emb + (size_t)(n0 + l16) * EDIM + 2 * hl;
    const float* brow = emb + (size_t)(m0 + l16) * EDIM + 2 * hl;
#pragma unroll
    for (int k0 = 0; k0 < EDIM; k0 += 4) {
        v2f a  = *(const v2f*)(arow + k0);
        v2f bm = *(const v2f*)(brow + k0);
        c = __builtin_amdgcn_wmma_f32_16x16x4_f32(false, a, false, bm,
                                                  (short)0, c, false, false);
    }
    return c;
}

// ============ Kernel 0: split x into bf16 hi/lo, pre-transposed xT[b][d][m] ============
// One block = 32 consecutive m rows for one batch. lane = m_loc (coalesced 64B b16
// stores, full lines written); d looped per wave (reads served by L0 due to reuse).
__global__ void __launch_bounds__(256, 1)
xsplit_kernel(const float* __restrict__ x, u16* __restrict__ xh, u16* __restrict__ xl) {
    const int t = threadIdx.x;
    const int lane = t & 31;   // m within tile
    const int wg   = t >> 5;   // d base
    const int m0 = blockIdx.x * 32;
    const int b  = blockIdx.y;
    const float* src = x + ((size_t)b * N_NODES + m0 + lane) * DIN;
    const size_t dstb = (size_t)b * DIN * (size_t)N_NODES + (size_t)(m0 + lane);
#pragma unroll
    for (int i = 0; i < 32; ++i) {
        const int d = wg + 8 * i;
        float v = src[d];
        u16 hi = f2bf(v);
        u16 lo = f2bf(v - bf2f(hi));
        xh[dstb + (size_t)d * N_NODES] = hi;
        xl[dstb + (size_t)d * N_NODES] = lo;
    }
}

// ======================= Kernel 1: adj = softmax(relu(E E^T)) =======================
__global__ void __launch_bounds__(256, 1)
adj_softmax_kernel(const float* __restrict__ emb,
                   u16* __restrict__ adj_hi, u16* __restrict__ adj_lo) {
    __shared__ float rowsum[16];
    const int t = threadIdx.x;
    const int w = t >> 5, lane = t & 31, l16 = lane & 15, hl = lane >> 4;
    const int n0 = blockIdx.x * 16;

    if (t < 16) rowsum[t] = 0.0f;
    __syncthreads();

    // -------- pass 1: accumulate exp-rowsums --------
    float racc[8] = {0.f, 0.f, 0.f, 0.f, 0.f, 0.f, 0.f, 0.f};
    for (int i = 0; i < 32; ++i) {
        const int m0 = (w * 32 + i) * 16;
        v8f c = logits_tile(emb, n0, m0, l16, hl);
#pragma unroll
        for (int v = 0; v < 8; ++v)
            racc[v] += __expf(fmaxf(c[v], 0.0f));
    }
#pragma unroll
    for (int v = 0; v < 8; ++v) {
        float s = racc[v];
        s += __shfl_xor(s, 1, 32);
        s += __shfl_xor(s, 2, 32);
        s += __shfl_xor(s, 4, 32);
        s += __shfl_xor(s, 8, 32);          // reduced within each 16-lane half
        if (l16 == 0) atomicAdd(&rowsum[v + 8 * hl], s);
    }
    __syncthreads();

    float rinv[8];
#pragma unroll
    for (int v = 0; v < 8; ++v) rinv[v] = 1.0f / rowsum[v + 8 * hl];

    // -------- pass 2: recompute, normalize, split-store bf16 hi/lo --------
    for (int i = 0; i < 32; ++i) {
        const int m0 = (w * 32 + i) * 16;
        v8f c = logits_tile(emb, n0, m0, l16, hl);
#pragma unroll
        for (int v = 0; v < 8; ++v) {
            float val = __expf(fmaxf(c[v], 0.0f)) * rinv[v];
            u16 hi = f2bf(val);
            u16 lo = f2bf(val - bf2f(hi));
            size_t idx = (size_t)(n0 + v + 8 * hl) * N_NODES + (size_t)(m0 + l16);
            adj_hi[idx] = hi;
            adj_lo[idx] = lo;
        }
    }
}

// ============ Kernel 2: out = relu((adj @ x) @ W + b), fused, streaming ============
// Block = (16-row n-tile, batch). Phase 1: pure global->WMMA stream, no LDS:
// per k-step/wave: 12x global_load_b128 + 6x v_wmma_f32_16x16x32_bf16
// (split product AhBh + AhBl + AlBh). Phase 2: h via LDS, exact f32 WMMA x W.
#define H_STRIDE 260   // floats; 4-bank skew per row -> conflict-free b64 reads

__global__ void __launch_bounds__(256, 1)
gcn_fused_kernel(const u16* __restrict__ adj_hi, const u16* __restrict__ adj_lo,
                 const u16* __restrict__ xh, const u16* __restrict__ xl,
                 const float* __restrict__ W, const float* __restrict__ bias,
                 float* __restrict__ out) {
    __shared__ __align__(16) float h_lds[16 * H_STRIDE];

    const int t = threadIdx.x;
    const int w = t >> 5, lane = t & 31, l16 = lane & 15, hl = lane >> 4;
    const int n0 = blockIdx.x * 16;
    const int b  = blockIdx.y;
    const int d0 = w * 32;                    // this wave's 32-wide D slice

    // Per-lane stream pointers (A: 16-bit A 16x32 layout; B: 16-bit B 32x16 layout)
    const u16* arow_h = adj_hi + (size_t)(n0 + l16) * N_NODES + 16 * hl;
    const u16* arow_l = adj_lo + (size_t)(n0 + l16) * N_NODES + 16 * hl;
    const size_t xrow = ((size_t)b * DIN + d0 + l16) * (size_t)N_NODES + 16 * hl;
    const u16* brow_h0 = xh + xrow;
    const u16* brow_l0 = xl + xrow;
    const u16* brow_h1 = brow_h0 + (size_t)16 * N_NODES;
    const u16* brow_l1 = brow_l0 + (size_t)16 * N_NODES;

    v8f acc0 = {}, acc1 = {};
    for (int m0 = 0; m0 < N_NODES; m0 += 32) {
        v16bf a_hi = ldfrag(arow_h + m0, 8);
        v16bf a_lo = ldfrag(arow_l + m0, 8);
        v16bf b_h0 = ldfrag(brow_h0 + m0, 8);
        v16bf b_l0 = ldfrag(brow_l0 + m0, 8);
        v16bf b_h1 = ldfrag(brow_h1 + m0, 8);
        v16bf b_l1 = ldfrag(brow_l1 + m0, 8);

        // stream-ahead hints for the per-lane adj rows
        __builtin_prefetch(arow_h + m0 + 256, 0, 1);
        __builtin_prefetch(arow_l + m0 + 256, 0, 1);

        acc0 = __builtin_amdgcn_wmma_f32_16x16x32_bf16(
            false, a_hi, false, b_h0, (short)0, acc0, false, false);
        acc0 = __builtin_amdgcn_wmma_f32_16x16x32_bf16(
            false, a_hi, false, b_l0, (short)0, acc0, false, false);
        acc0 = __builtin_amdgcn_wmma_f32_16x16x32_bf16(
            false, a_lo, false, b_h0, (short)0, acc0, false, false);

        acc1 = __builtin_amdgcn_wmma_f32_16x16x32_bf16(
            false, a_hi, false, b_h1, (short)0, acc1, false, false);
        acc1 = __builtin_amdgcn_wmma_f32_16x16x32_bf16(
            false, a_hi, false, b_l1, (short)0, acc1, false, false);
        acc1 = __builtin_amdgcn_wmma_f32_16x16x32_bf16(
            false, a_lo, false, b_h1, (short)0, acc1, false, false);
    }

    // ---- spill h band to LDS (C/D layout: VGPR v -> row v + 8*hl, col = lane%16) ----
#pragma unroll
    for (int v = 0; v < 8; ++v) {
        h_lds[(v + 8 * hl) * H_STRIDE + d0 + l16]      = acc0[v];
        h_lds[(v + 8 * hl) * H_STRIDE + d0 + 16 + l16] = acc1[v];
    }
    __syncthreads();

    // ---- stage C: out = relu(h @ W + bias), exact f32 WMMA 16x16x4 ----
    v8f oacc[2] = {};
#pragma unroll 8
    for (int k0 = 0; k0 < DIN; k0 += 4) {
        v2f a = *(const v2f*)(&h_lds[l16 * H_STRIDE + k0 + 2 * hl]);
#pragma unroll
        for (int tt = 0; tt < 2; ++tt) {
            const int o = d0 + tt * 16 + l16;
            v2f bw;
            bw.x = W[(size_t)(k0 + 2 * hl) * DOUT + o];
            bw.y = W[(size_t)(k0 + 2 * hl + 1) * DOUT + o];
            oacc[tt] = __builtin_amdgcn_wmma_f32_16x16x4_f32(
                false, a, false, bw, (short)0, oacc[tt], false, false);
        }
    }
#pragma unroll
    for (int tt = 0; tt < 2; ++tt) {
        const int o = d0 + tt * 16 + l16;
        const float bv = bias[o];
#pragma unroll
        for (int v = 0; v < 8; ++v) {
            float r = fmaxf(oacc[tt][v] + bv, 0.0f);
            out[((size_t)b * N_NODES + n0 + v + 8 * hl) * DOUT + o] = r;
        }
    }
}

// ======================= host-side launcher =======================
extern "C" void kernel_launch(void* const* d_in, const int* in_sizes, int n_in,
                              void* d_out, int out_size, void* d_ws, size_t ws_size,
                              hipStream_t stream) {
    // inputs: 0=x [32,4096,256] f32, 1=node_ids (identity arange, unused),
    //         2=emb_table [4096,64] f32, 3=W [256,256] f32, 4=b [256] f32
    const float* x    = (const float*)d_in[0];
    const float* emb  = (const float*)d_in[2];
    const float* W    = (const float*)d_in[3];
    const float* bias = (const float*)d_in[4];
    float* out = (float*)d_out;

    // workspace layout (u16 elements):
    //   adj_hi, adj_lo : N*N each          (32 MB each)
    //   xh, xl         : B*DIN*N each      (64 MB each)  -> 192 MB total
    const size_t NN = (size_t)N_NODES * N_NODES;
    const size_t XT = (size_t)BATCH * DIN * N_NODES;
    u16* adj_hi = (u16*)d_ws;
    u16* adj_lo = adj_hi + NN;
    u16* xhp    = adj_lo + NN;
    u16* xlp    = xhp + XT;

    dim3 gx(N_NODES / 32, BATCH);
    xsplit_kernel<<<gx, 256, 0, stream>>>(x, xhp, xlp);

    adj_softmax_kernel<<<N_NODES / 16, 256, 0, stream>>>(emb, adj_hi, adj_lo);

    dim3 grid(N_NODES / 16, BATCH);
    gcn_fused_kernel<<<grid, 256, 0, stream>>>(adj_hi, adj_lo, xhp, xlp, W, bias, out);
}
